// MultiHeadAttention_35837207117857
// MI455X (gfx1250) — matrix operand order repaired and checked by reading, not compiled, not run on hardware
//
#include <hip/hip_runtime.h>
#include <hip/hip_bf16.h>

// ---------- types ----------
typedef _Float16 h4  __attribute__((ext_vector_type(4)));
typedef _Float16 h8  __attribute__((ext_vector_type(8)));
typedef _Float16 v16h __attribute__((ext_vector_type(16)));
typedef float    v8f  __attribute__((ext_vector_type(8)));

#define NB 4
#define NH 16
#define SEQ 2048
#define DH 64
#define DM 1024
#define GM (NB*SEQ)   // 8192 rows for all projections

__device__ __forceinline__ v8f wmma16(v16h a, v16h b, v8f c) {
  return __builtin_amdgcn_wmma_f32_16x16x32_f16(false, a, false, b, (short)0, c, false, false);
}

// CDNA5 async global->LDS copy (16B per lane), tracked by ASYNCcnt.
__device__ __forceinline__ void async_load_b128(const void* gptr, void* lptr) {
  unsigned lds = (unsigned)(size_t)lptr;                 // low 32 bits = LDS offset
  unsigned long long g = (unsigned long long)(size_t)gptr;
  asm volatile("global_load_async_to_lds_b128 %0, %1, off"
               :: "v"(lds), "v"(g) : "memory");
}
__device__ __forceinline__ void wait_async0() {
  asm volatile("s_wait_asynccnt 0" ::: "memory");
}

// A-fragment (16x32 f16): lane(0-15)=row, per-lane K = {g*8..g*8+7, 16+g*8..+7}, g=lane/16
__device__ __forceinline__ v16h load_a_frag(const _Float16* base, int stride, int row, int kbase) {
  const int lane = threadIdx.x & 31;
  const _Float16* p = base + (size_t)(row + (lane & 15)) * stride + kbase + ((lane >> 4) << 3);
  h8 lo = *(const h8*)p;
  h8 hi = *(const h8*)(p + 16);
  return __builtin_shufflevector(lo, hi, 0,1,2,3,4,5,6,7,8,9,10,11,12,13,14,15);
}

// B-fragment (32x16 f16): lane(0-15)=col N, per-lane K = {g*8..g*8+7, 16+g*8..+7}, g=lane/16
__device__ __forceinline__ v16h load_b_frag(const _Float16* base, int stride, int row, int kbase) {
  const int lane = threadIdx.x & 31;
  const _Float16* p = base + (size_t)(row + (lane & 15)) * stride + kbase + ((lane >> 4) << 3);
  h8 lo = *(const h8*)p;
  h8 hi = *(const h8*)(p + 16);
  return __builtin_shufflevector(lo, hi, 0,1,2,3,4,5,6,7,8,9,10,11,12,13,14,15);
}

// ---------------- GEMM: C[8192 x 1024] = A[8192 x 1024] @ W[1024 x 1024] + bias ----------------
// mode 0: A = f16 (Z), out = f32 row-major (d_out)
// mode 1: A = f32, out = f16 scaled 1/8, layout [B][H][S][D]   (Q)
// mode 2: A = f32, out = f16, layout [B][H][S][D]              (K)
// mode 3: A = f32, out = f16, layout [B][H][D][S]              (V transposed)
__global__ __launch_bounds__(256) void mha_gemm_kernel(
    const float* __restrict__ A32, const _Float16* __restrict__ A16,
    const float* __restrict__ W, const float* __restrict__ bias,
    float* __restrict__ Cf, _Float16* __restrict__ Ch, int mode)
{
  __shared__ __attribute__((aligned(16))) _Float16 As[2][128 * 40];
  __shared__ __attribute__((aligned(16))) _Float16 Bs[2][64 * 40];
  const int tid  = threadIdx.x;
  const int lane = tid & 31;
  const int w    = tid >> 5;
  const int wm   = w & 3, wn = w >> 2;
  const int m0   = blockIdx.x * 128;
  const int n0   = blockIdx.y * 64;

  v8f acc[2][2] = {};
  float4 ra[4];         // staged A (f32 modes)
  float4 rw[2];         // staged W

  // issue global fetches for K-tile starting at k0; async A16 goes straight to LDS buf
  auto issue_tile = [&](int k0, int buf) {
    if (mode == 0) {
      for (int i = 0; i < 2; ++i) {
        int c = tid + i * 256;                       // 512 x 16B
        int row = c >> 2, c8 = (c & 3) << 3;
        async_load_b128(A16 + (size_t)(m0 + row) * DM + k0 + c8,
                        &As[buf][row * 40 + c8]);
      }
    } else {
      for (int i = 0; i < 4; ++i) {
        int c = tid + i * 256;                       // 1024 x float4
        int row = c >> 3, c4 = (c & 7) << 2;
        ra[i] = *(const float4*)(A32 + (size_t)(m0 + row) * DM + k0 + c4);
      }
    }
    for (int i = 0; i < 2; ++i) {
      int c = tid + i * 256;                         // 512 x float4 (32x64 W tile)
      int kr = c >> 4, nc = (c & 15) << 2;
      rw[i] = *(const float4*)(W + (size_t)(k0 + kr) * DM + n0 + nc);
    }
  };

  issue_tile(0, 0);
  const int NT = DM / 32;
  for (int kt = 0; kt < NT; ++kt) {
    const int b = kt & 1;
    // drain staged registers -> LDS buf b (compiler inserts single s_wait_loadcnt)
    if (mode != 0) {
      for (int i = 0; i < 4; ++i) {
        int c = tid + i * 256;
        int row = c >> 3, c4 = (c & 7) << 2;
        h4 hv; hv.x = (_Float16)ra[i].x; hv.y = (_Float16)ra[i].y;
        hv.z = (_Float16)ra[i].z; hv.w = (_Float16)ra[i].w;
        *(h4*)(&As[b][row * 40 + c4]) = hv;
      }
    }
    for (int i = 0; i < 2; ++i) {
      int c = tid + i * 256;
      int kr = c >> 4, nc = (c & 15) << 2;
      Bs[b][(nc + 0) * 40 + kr] = (_Float16)rw[i].x;
      Bs[b][(nc + 1) * 40 + kr] = (_Float16)rw[i].y;
      Bs[b][(nc + 2) * 40 + kr] = (_Float16)rw[i].z;
      Bs[b][(nc + 3) * 40 + kr] = (_Float16)rw[i].w;
    }
    if (mode == 0) wait_async0();
    __syncthreads();
    if (kt + 1 < NT) issue_tile((kt + 1) * 32, 1 - b);   // overlap with WMMAs below

    v16h a0 = load_a_frag(As[b], 40, wm * 32, 0);
    v16h a1 = load_a_frag(As[b], 40, wm * 32 + 16, 0);
    v16h b0 = load_b_frag(Bs[b], 40, wn * 32, 0);
    v16h b1 = load_b_frag(Bs[b], 40, wn * 32 + 16, 0);
    acc[0][0] = wmma16(a0, b0, acc[0][0]);
    acc[0][1] = wmma16(a0, b1, acc[0][1]);
    acc[1][0] = wmma16(a1, b0, acc[1][0]);
    acc[1][1] = wmma16(a1, b1, acc[1][1]);
  }

  const int hf = lane >> 4, nl = lane & 15;
  for (int tm = 0; tm < 2; ++tm)
    for (int tn = 0; tn < 2; ++tn) {
      int nglob = n0 + wn * 32 + tn * 16 + nl;
      int mbase = m0 + wm * 32 + tm * 16 + hf * 8;
      float bv = bias[nglob];
      for (int v = 0; v < 8; ++v) {
        int mg = mbase + v;
        float val = acc[tm][tn][v] + bv;
        if (mode == 0) {
          Cf[(size_t)mg * DM + nglob] = val;
        } else {
          int b = mg >> 11, s = mg & (SEQ - 1);
          int h = nglob >> 6, d = nglob & (DH - 1);
          if (mode == 1)
            Ch[(((size_t)b * NH + h) * SEQ + s) * DH + d] = (_Float16)(val * 0.125f);
          else if (mode == 2)
            Ch[(((size_t)b * NH + h) * SEQ + s) * DH + d] = (_Float16)val;
          else
            Ch[(((size_t)b * NH + h) * DH + d) * SEQ + s] = (_Float16)val;
        }
      }
    }
}

// ---------------- Flash attention (causal), per (b,h): 128 q-rows per block ----------------
__global__ __launch_bounds__(256) void mha_attn_kernel(
    const _Float16* __restrict__ Q, const _Float16* __restrict__ K,
    const _Float16* __restrict__ Vt, _Float16* __restrict__ Z)
{
  __shared__ __attribute__((aligned(16))) _Float16 Qs[128 * 72];
  __shared__ __attribute__((aligned(16))) _Float16 Ks[2][64 * 72];
  __shared__ __attribute__((aligned(16))) _Float16 Vts[2][64 * 72];
  __shared__ __attribute__((aligned(16))) _Float16 Ps[8 * 16 * 72];

  const int tid = threadIdx.x, lane = tid & 31, w = tid >> 5;
  const int qb = blockIdx.x & 15;      // 16 query blocks of 128
  const int bh = blockIdx.x >> 4;      // 0..63 = b*16+h
  const int q0 = qb * 128;
  const _Float16* Qg = Q + (size_t)bh * SEQ * DH;
  const _Float16* Kg = K + (size_t)bh * SEQ * DH;
  const _Float16* Vg = Vt + (size_t)bh * DH * SEQ;

  auto issue_kv = [&](int kbase, int buf) {
    for (int i = 0; i < 2; ++i) {                    // 512 x 16B each
      int c = tid + i * 256;
      int row = c >> 3, c8 = (c & 7) << 3;
      async_load_b128(Kg + (size_t)(kbase + row) * DH + c8, &Ks[buf][row * 72 + c8]);
      async_load_b128(Vg + (size_t)row * SEQ + kbase + c8, &Vts[buf][row * 72 + c8]);
    }
  };

  // Q tile (128x64) + first K/V block, all async
  for (int i = 0; i < 4; ++i) {
    int c = tid + i * 256;
    int row = c >> 3, c8 = (c & 7) << 3;
    async_load_b128(Qg + (size_t)(q0 + row) * DH + c8, &Qs[row * 72 + c8]);
  }
  issue_kv(0, 0);
  wait_async0();
  __syncthreads();

  v16h qf0 = load_a_frag(Qs, 72, w * 16, 0);
  v16h qf1 = load_a_frag(Qs, 72, w * 16, 32);

  v8f O[4] = {};
  float rm[8], rl[8];
  for (int v = 0; v < 8; ++v) { rm[v] = -3.0e38f; rl[v] = 0.0f; }

  _Float16* Ph = Ps + w * 16 * 72;
  const int hf = lane >> 4, nl = lane & 15;
  const int nkb = qb * 2 + 2;          // key blocks up to causal diagonal

  for (int kb = 0; kb < nkb; ++kb) {
    const int kbase = kb * 64;
    const int bsel = kb & 1;
    if (kb + 1 < nkb) issue_kv((kb + 1) * 64, 1 - bsel);   // prefetch next block

    // S strip 16x64 = 4 tiles, each 2 WMMAs over d
    v8f s[4];
    for (int j = 0; j < 4; ++j) {
      v8f c0 = {};
      v16h bk0 = load_b_frag(Ks[bsel], 72, j * 16, 0);
      v16h bk1 = load_b_frag(Ks[bsel], 72, j * 16, 32);
      c0 = wmma16(qf0, bk0, c0);
      c0 = wmma16(qf1, bk1, c0);
      s[j] = c0;
    }

    // causal mask + row max
    float tmax[8];
    for (int v = 0; v < 8; ++v) tmax[v] = -3.0e38f;
    for (int j = 0; j < 4; ++j) {
      int ncol = kbase + j * 16 + nl;
      for (int v = 0; v < 8; ++v) {
        int mrow = q0 + w * 16 + hf * 8 + v;
        float sv = s[j][v];
        if (ncol > mrow) sv = -3.0e38f;
        s[j][v] = sv;
        tmax[v] = fmaxf(tmax[v], sv);
      }
    }
    for (int m = 1; m < 16; m <<= 1)
      for (int v = 0; v < 8; ++v) tmax[v] = fmaxf(tmax[v], __shfl_xor(tmax[v], m, 32));

    float alpha[8], rs[8];
    for (int v = 0; v < 8; ++v) {
      float mn = fmaxf(rm[v], tmax[v]);
      alpha[v] = __expf(rm[v] - mn);
      rm[v] = mn;
      rs[v] = 0.0f;
    }

    // P = exp(S - m); row sums; write P strip to LDS (re-swizzle C-layout -> A-frag layout)
    for (int j = 0; j < 4; ++j)
      for (int v = 0; v < 8; ++v) {
        float p = __expf(s[j][v] - rm[v]);
        rs[v] += p;
        Ph[(size_t)(hf * 8 + v) * 72 + j * 16 + nl] = (_Float16)p;
      }
    for (int m = 1; m < 16; m <<= 1)
      for (int v = 0; v < 8; ++v) rs[v] += __shfl_xor(rs[v], m, 32);
    for (int v = 0; v < 8; ++v) rl[v] = rl[v] * alpha[v] + rs[v];
    for (int t = 0; t < 4; ++t)
      for (int v = 0; v < 8; ++v) O[t][v] *= alpha[v];

    asm volatile("s_wait_dscnt 0" ::: "memory");   // wave-local P RAW through LDS

    v16h pf0 = load_a_frag(Ph, 72, 0, 0);
    v16h pf1 = load_a_frag(Ph, 72, 0, 32);
    for (int t = 0; t < 4; ++t) {
      v16h bv0 = load_b_frag(Vts[bsel], 72, t * 16, 0);
      v16h bv1 = load_b_frag(Vts[bsel], 72, t * 16, 32);
      O[t] = wmma16(pf0, bv0, O[t]);
      O[t] = wmma16(pf1, bv1, O[t]);
    }

    if (kb + 1 < nkb) {
      wait_async0();       // prefetched K/V landed
      __syncthreads();     // everyone done reading buf bsel before it is overwritten
    }
  }

  // epilogue: normalize and write Z as f16 [B][S][H*D]
  int b = bh >> 4, h = bh & 15;
  for (int v = 0; v < 8; ++v) rl[v] = 1.0f / rl[v];
  for (int t = 0; t < 4; ++t)
    for (int v = 0; v < 8; ++v) {
      int mg = q0 + w * 16 + hf * 8 + v;
      int d  = t * 16 + nl;
      Z[((size_t)b * SEQ + mg) * DM + h * DH + d] = (_Float16)(O[t][v] * rl[v]);
    }
}

// ---------------- launch ----------------
extern "C" void kernel_launch(void* const* d_in, const int* in_sizes, int n_in,
                              void* d_out, int out_size, void* d_ws, size_t ws_size,
                              hipStream_t stream) {
  (void)in_sizes; (void)n_in; (void)out_size; (void)ws_size;
  const float* data    = (const float*)d_in[0];
  const float* context = (const float*)d_in[1];
  const float* wq = (const float*)d_in[2]; const float* bq = (const float*)d_in[3];
  const float* wk = (const float*)d_in[4]; const float* bk = (const float*)d_in[5];
  const float* wv = (const float*)d_in[6]; const float* bv = (const float*)d_in[7];
  const float* wo = (const float*)d_in[8]; const float* bo = (const float*)d_in[9];
  float* out = (float*)d_out;

  // workspace: Q, K, Vt, Z each B*S*1024 f16 = 16 MB -> 64 MB total
  char* ws = (char*)d_ws;
  const size_t SEG = (size_t)NB * NH * SEQ * DH * sizeof(_Float16);
  _Float16* qws = (_Float16*)(ws);
  _Float16* kws = (_Float16*)(ws + SEG);
  _Float16* vws = (_Float16*)(ws + 2 * SEG);
  _Float16* zws = (_Float16*)(ws + 3 * SEG);

  dim3 gg(GM / 128, DM / 64), blk(256);
  mha_gemm_kernel<<<gg, blk, 0, stream>>>(data,    nullptr, wq, bq, nullptr, qws, 1);
  mha_gemm_kernel<<<gg, blk, 0, stream>>>(context, nullptr, wk, bk, nullptr, kws, 2);
  mha_gemm_kernel<<<gg, blk, 0, stream>>>(context, nullptr, wv, bv, nullptr, vws, 3);
  mha_attn_kernel<<<NB * NH * (SEQ / 128), blk, 0, stream>>>(qws, kws, vws, zws);
  mha_gemm_kernel<<<gg, blk, 0, stream>>>(nullptr, zws, wo, bo, out, nullptr, 0);
}
